// GRU_Module_76201309766404
// MI455X (gfx1250) — matrix operand order repaired and checked by reading, not compiled, not run on hardware
//
#include <hip/hip_runtime.h>
#include <cstdint>

// ---------------------------------------------------------------------------
// ConvGRU (B=8, C=32, T=32, H=64, W=64, R=32) for gfx1250 (MI455X, wave32).
//
// Step t:
//   kernel A: [x_t | h] --(3x3 conv, K=576, N=64 via bf16 WMMA)--> (u, r) preacts
//             upd = sigmoid(.), rst = sigmoid(.), writes:
//               hrT   = bf16(h * rst)            (channels-last, conv input for B)
//               base  = h * (1 - upd)            (fp32)
//               updw  = upd                      (fp32)
//   kernel B: [x_t | hr] --(3x3 conv, K=576, N=32)--> out-gate preact
//             h_new = base + tanh(.) * upd ; writes hT (bf16) + out (fp32 NCTHW)
//
// Staging uses GLOBAL_LOAD_ASYNC_TO_LDS_B128 (ASYNCcnt) when available.
// ---------------------------------------------------------------------------

typedef __attribute__((ext_vector_type(16))) __bf16 v16bf;
typedef __attribute__((ext_vector_type(8)))  float  v8f;

union Frag {
    v16bf v;
    uint4 q[2];
};

#define CB   8      // batch
#define CC   32     // input channels
#define CT   32     // frames
#define CH   64
#define CW   64
#define CHW  4096
#define CR   32     // recurrent channels
#define STG_C 72    // 64 channels + 8 pad (144B pitch -> conflict-free b128)
#define STG_P 264   // 4 rows x 66 cols (halo)

#if defined(__HIP_DEVICE_COMPILE__) && __has_builtin(__builtin_amdgcn_global_load_async_to_lds_b128)
#define USE_ASYNC_STAGE 1
#else
#define USE_ASYNC_STAGE 0
#endif

#if USE_ASYNC_STAGE
// Match the builtin's parameter types exactly:
//   arg0: int __attribute__((vector_size(16))) __device__ * (addrspace 1)
//   arg1: same pointee in addrspace 3 (LDS)
typedef int v4i_vs __attribute__((vector_size(16)));
typedef __attribute__((address_space(1))) v4i_vs gv4i;
typedef __attribute__((address_space(3))) v4i_vs lv4i;
#endif

#if defined(__HIP_DEVICE_COMPILE__)
__device__ __forceinline__ void wait_async_zero()
{
#if __has_builtin(__builtin_amdgcn_s_wait_asynccnt)
    __builtin_amdgcn_s_wait_asynccnt(0);
#elif USE_ASYNC_STAGE
    asm volatile("s_wait_asynccnt 0x0" ::: "memory");
#endif
}
#else
__device__ __forceinline__ void wait_async_zero() {}
#endif

// ---------------------------------------------------------------------------
// Phase 0a: x (B,C,T,H,W) fp32  ->  xT (B,T,H,W,C) bf16 (channels-last)
// ---------------------------------------------------------------------------
__global__ __launch_bounds__(256) void convgru_transpose_x(
    const float* __restrict__ x, __bf16* __restrict__ xT)
{
    __shared__ float tile[CC][CW + 1];
    const int bx = blockIdx.x;              // B*T*H blocks
    const int b  = bx >> 11;                // / (T*H)
    const int rm = bx & 2047;
    const int t  = rm >> 6;
    const int hh = rm & 63;
    const int tid = threadIdx.x;

    for (int k = tid; k < CC * CW; k += 256) {
        int c = k >> 6, w = k & 63;
        tile[c][w] = x[(((size_t)b * CC + c) * CT + t) * CHW + hh * CW + w];
    }
    __syncthreads();
    for (int k = tid; k < CC * CW; k += 256) {
        int w = k >> 5, c = k & 31;
        xT[(((size_t)b * CT + t) * CHW + hh * CW + w) * CC + c] = (__bf16)tile[c][w];
    }
}

// ---------------------------------------------------------------------------
// Phase 0b: prepack weights to bf16 [n][tap(18)][c(32)] and fused biases.
//   WA: n 0..31 = update gate (ui|uh taps), n 32..63 = reset gate (ri|rh)
//   WB: n 0..31 = out gate (oi|oh)
// ---------------------------------------------------------------------------
__global__ __launch_bounds__(256) void convgru_prepack(
    const float* __restrict__ w_ri, const float* __restrict__ b_ri,
    const float* __restrict__ w_rh, const float* __restrict__ b_rh,
    const float* __restrict__ w_ui, const float* __restrict__ b_ui,
    const float* __restrict__ w_uh, const float* __restrict__ b_uh,
    const float* __restrict__ w_oi, const float* __restrict__ b_oi,
    const float* __restrict__ w_oh, const float* __restrict__ b_oh,
    __bf16* __restrict__ WA, __bf16* __restrict__ WB,
    float* __restrict__ bA, float* __restrict__ bB)
{
    const int tid = threadIdx.x;
    for (int idx = tid; idx < 64 * 18 * 32; idx += 256) {
        int n   = idx / 576;
        int tap = (idx / 32) % 18;
        int c   = idx & 31;
        int gate = n >> 5, ch = n & 31;
        float wv;
        if (tap < 9)  wv = (gate == 0 ? w_ui : w_ri)[(ch * 32 + c) * 9 + tap];
        else          wv = (gate == 0 ? w_uh : w_rh)[(ch * 32 + c) * 9 + (tap - 9)];
        WA[idx] = (__bf16)wv;
    }
    for (int idx = tid; idx < 32 * 18 * 32; idx += 256) {
        int n   = idx / 576;
        int tap = (idx / 32) % 18;
        int c   = idx & 31;
        float wv;
        if (tap < 9)  wv = w_oi[(n * 32 + c) * 9 + tap];
        else          wv = w_oh[(n * 32 + c) * 9 + (tap - 9)];
        WB[idx] = (__bf16)wv;
    }
    if (tid < 64) {
        int gate = tid >> 5, ch = tid & 31;
        bA[tid] = gate ? (b_ri[ch] + b_rh[ch]) : (b_ui[ch] + b_uh[ch]);
    }
    if (tid < 32) bB[tid] = b_oi[tid] + b_oh[tid];
}

// ---------------------------------------------------------------------------
// Shared staging: 4 rows x 66 cols, 64 channels bf16 ([x | other]) per pos.
// In-range lanes: async global->LDS b128 copies (no VGPR round trip).
// Halo lanes: zero-fill via normal ds stores.
// ---------------------------------------------------------------------------
__device__ __forceinline__ void stage_tile(
    __bf16* stage, const __bf16* __restrict__ srcX, const __bf16* __restrict__ srcH,
    int row0, int tid)
{
    for (int idx = tid; idx < STG_P * 8; idx += 256) {
        int p = idx >> 3, q = idx & 7;
        int pr = p / 66, pc = p - pr * 66;
        int in_r = row0 - 1 + pr, in_w = pc - 1;
        __bf16* dst = stage + p * STG_C + ((q < 4) ? (q * 8) : (32 + (q - 4) * 8));
        if ((unsigned)in_r < 64u && (unsigned)in_w < 64u) {
            const __bf16* src = ((q < 4) ? srcX : srcH)
                              + (size_t)(in_r * CW + in_w) * 32 + (size_t)(q & 3) * 8;
#if USE_ASYNC_STAGE
            __builtin_amdgcn_global_load_async_to_lds_b128(
                (gv4i*)src, (lv4i*)dst, 0, 0);
#else
            *(uint4*)dst = *(const uint4*)src;
#endif
        } else {
            *(uint4*)dst = make_uint4(0u, 0u, 0u, 0u);
        }
    }
#if USE_ASYNC_STAGE
    wait_async_zero();
#endif
}

__device__ __forceinline__ void load_afrag(
    Frag& af, const __bf16* stage, int m, int kh, int kw, int sect, int lh)
{
    int p = ((m >> 6) + kh) * 66 + (m & 63) + kw;
    af.q[0] = *(const uint4*)(stage + p * STG_C + sect + lh * 8);
    af.q[1] = *(const uint4*)(stage + p * STG_C + sect + 16 + lh * 8);
}

__device__ __forceinline__ void load_bfrag(
    Frag& bf, const __bf16* __restrict__ Wg, int n, int tap, int lh)
{
    const uint4* wp = (const uint4*)(Wg + ((n * 18 + tap) * 32 + lh * 8));
    bf.q[0] = wp[0];
    bf.q[1] = wp[2];
}

__device__ __forceinline__ float sigmoidf_(float v)
{
    return 1.0f / (1.0f + __expf(-v));
}

// ---------------------------------------------------------------------------
// Step kernel A: update + reset gates (N=64), epilogue gating.
// grid: B * H/2 = 256 blocks of 256 threads (8 waves).
// ---------------------------------------------------------------------------
__global__ __launch_bounds__(256) void convgru_step_ur(
    const __bf16* __restrict__ xT, const __bf16* __restrict__ hT,
    __bf16* __restrict__ hrT, float* __restrict__ updw, float* __restrict__ basew,
    const __bf16* __restrict__ WA, const float* __restrict__ bA, int t)
{
    __shared__ __align__(16) __bf16 stage[STG_P * STG_C];
    const int tid  = threadIdx.x;
    const int b    = blockIdx.x >> 5;
    const int row0 = (blockIdx.x & 31) << 1;
    const __bf16* xsl = xT + ((size_t)(b * CT + t)) * CHW * CC;
    const __bf16* hsl = hT + (size_t)b * CHW * CR;

    stage_tile(stage, xsl, hsl, row0, tid);
    __syncthreads();

    const int lane = tid & 31, wave = tid >> 5;
    const int lm = lane & 15, lh = lane >> 4;
    const int mblk  = wave & 3;   // 32 M-rows
    const int nhalf = wave >> 2;  // 0: update, 1: reset

    v8f acc[2][2];
    const v8f vz = {0.f, 0.f, 0.f, 0.f, 0.f, 0.f, 0.f, 0.f};
    acc[0][0] = vz; acc[0][1] = vz; acc[1][0] = vz; acc[1][1] = vz;

    for (int tap = 0; tap < 18; ++tap) {
        int t9 = (tap >= 9) ? tap - 9 : tap;
        int kh = t9 / 3, kw = t9 - kh * 3;
        int sect = (tap >= 9) ? 32 : 0;
        Frag bf0, bf1;
        load_bfrag(bf0, WA, nhalf * 32 + lm,      tap, lh);
        load_bfrag(bf1, WA, nhalf * 32 + 16 + lm, tap, lh);
        for (int mt = 0; mt < 2; ++mt) {
            Frag af;
            load_afrag(af, stage, mblk * 32 + mt * 16 + lm, kh, kw, sect, lh);
            acc[mt][0] = __builtin_amdgcn_wmma_f32_16x16x32_bf16(
                false, af.v, false, bf0.v, (short)0, acc[mt][0], false, false);
            acc[mt][1] = __builtin_amdgcn_wmma_f32_16x16x32_bf16(
                false, af.v, false, bf1.v, (short)0, acc[mt][1], false, false);
        }
    }

    for (int mt = 0; mt < 2; ++mt) {
        for (int nt = 0; nt < 2; ++nt) {
            int chan = nt * 16 + lm;
            float bias = bA[nhalf * 32 + chan];
            for (int v = 0; v < 8; ++v) {
                int m = mblk * 32 + mt * 16 + lh * 8 + v;
                float s = sigmoidf_(acc[mt][nt][v] + bias);
                int pcen = ((m >> 6) + 1) * 66 + (m & 63) + 1;
                float hval = (float)stage[pcen * STG_C + 32 + chan];
                size_t fl = (size_t)b * CHW + (size_t)(row0 * CW + m);
                if (nhalf == 0) {
                    updw [fl * 32 + chan] = s;
                    basew[fl * 32 + chan] = hval * (1.0f - s);
                } else {
                    hrT[fl * 32 + chan] = (__bf16)(hval * s);
                }
            }
        }
    }
}

// ---------------------------------------------------------------------------
// Step kernel B: out gate (N=32) + h update; writes hT and output slab.
// ---------------------------------------------------------------------------
__global__ __launch_bounds__(256) void convgru_step_o(
    const __bf16* __restrict__ xT, const __bf16* __restrict__ hrT,
    const float* __restrict__ updw, const float* __restrict__ basew,
    const __bf16* __restrict__ WB, const float* __restrict__ bB,
    __bf16* __restrict__ hT, float* __restrict__ out, int t)
{
    __shared__ __align__(16) __bf16 stage[STG_P * STG_C];
    const int tid  = threadIdx.x;
    const int b    = blockIdx.x >> 5;
    const int row0 = (blockIdx.x & 31) << 1;
    const __bf16* xsl  = xT  + ((size_t)(b * CT + t)) * CHW * CC;
    const __bf16* hrsl = hrT + (size_t)b * CHW * CR;

    stage_tile(stage, xsl, hrsl, row0, tid);
    __syncthreads();

    const int lane = tid & 31, wave = tid >> 5;
    const int lm = lane & 15, lh = lane >> 4;

    v8f acc[2];
    const v8f vz = {0.f, 0.f, 0.f, 0.f, 0.f, 0.f, 0.f, 0.f};
    acc[0] = vz; acc[1] = vz;

    for (int tap = 0; tap < 18; ++tap) {
        int t9 = (tap >= 9) ? tap - 9 : tap;
        int kh = t9 / 3, kw = t9 - kh * 3;
        int sect = (tap >= 9) ? 32 : 0;
        Frag bf0, bf1, af;
        load_bfrag(bf0, WB, lm,      tap, lh);
        load_bfrag(bf1, WB, 16 + lm, tap, lh);
        load_afrag(af, stage, wave * 16 + lm, kh, kw, sect, lh);
        acc[0] = __builtin_amdgcn_wmma_f32_16x16x32_bf16(
            false, af.v, false, bf0.v, (short)0, acc[0], false, false);
        acc[1] = __builtin_amdgcn_wmma_f32_16x16x32_bf16(
            false, af.v, false, bf1.v, (short)0, acc[1], false, false);
    }

    for (int nt = 0; nt < 2; ++nt) {
        int chan = nt * 16 + lm;
        float bias = bB[chan];
        float hn[8];
        for (int v = 0; v < 8; ++v) {
            int m = wave * 16 + lh * 8 + v;
            float htg = tanhf(acc[nt][v] + bias);
            size_t fl = (size_t)b * CHW + (size_t)(row0 * CW + m);
            float hnew = basew[fl * 32 + chan] + htg * updw[fl * 32 + chan];
            hT[fl * 32 + chan] = (__bf16)hnew;
            hn[v] = hnew;
        }
        size_t ob = (((size_t)b * CR + chan) * CT + t) * CHW
                  + (size_t)(row0 * CW + wave * 16 + lh * 8);
        *(float4*)(out + ob)     = make_float4(hn[0], hn[1], hn[2], hn[3]);
        *(float4*)(out + ob + 4) = make_float4(hn[4], hn[5], hn[6], hn[7]);
    }
}

// ---------------------------------------------------------------------------
// Host launcher
// ---------------------------------------------------------------------------
extern "C" void kernel_launch(void* const* d_in, const int* in_sizes, int n_in,
                              void* d_out, int out_size, void* d_ws, size_t ws_size,
                              hipStream_t stream)
{
    const float* x    = (const float*)d_in[0];
    const float* w_ri = (const float*)d_in[1];
    const float* b_ri = (const float*)d_in[2];
    const float* w_rh = (const float*)d_in[3];
    const float* b_rh = (const float*)d_in[4];
    const float* w_ui = (const float*)d_in[5];
    const float* b_ui = (const float*)d_in[6];
    const float* w_uh = (const float*)d_in[7];
    const float* b_uh = (const float*)d_in[8];
    const float* w_oi = (const float*)d_in[9];
    const float* b_oi = (const float*)d_in[10];
    const float* w_oh = (const float*)d_in[11];
    const float* b_oh = (const float*)d_in[12];
    float* out = (float*)d_out;

    char* ws = (char*)d_ws;
    const size_t XT_OFF   = 0;                         // 64 MiB bf16
    const size_t HT_OFF   = XT_OFF + (size_t)CB*CT*CHW*CC*2;   // 2 MiB
    const size_t HRT_OFF  = HT_OFF + (size_t)CB*CHW*CR*2;      // 2 MiB
    const size_t UPD_OFF  = HRT_OFF + (size_t)CB*CHW*CR*2;     // 4 MiB
    const size_t BASE_OFF = UPD_OFF + (size_t)CB*CHW*CR*4;     // 4 MiB
    const size_t WA_OFF   = BASE_OFF + (size_t)CB*CHW*CR*4;
    const size_t WB_OFF   = WA_OFF + 64*18*32*2 + 1024;
    const size_t BA_OFF   = WB_OFF + 32*18*32*2 + 1024;
    const size_t BB_OFF   = BA_OFF + 1024;

    __bf16* xT  = (__bf16*)(ws + XT_OFF);
    __bf16* hT  = (__bf16*)(ws + HT_OFF);
    __bf16* hrT = (__bf16*)(ws + HRT_OFF);
    float*  upd = (float*)(ws + UPD_OFF);
    float*  bas = (float*)(ws + BASE_OFF);
    __bf16* WA  = (__bf16*)(ws + WA_OFF);
    __bf16* WB  = (__bf16*)(ws + WB_OFF);
    float*  bA  = (float*)(ws + BA_OFF);
    float*  bB  = (float*)(ws + BB_OFF);

    (void)hipMemsetAsync(hT, 0, (size_t)CB * CHW * CR * 2, stream);
    convgru_transpose_x<<<CB * CT * CH, 256, 0, stream>>>(x, xT);
    convgru_prepack<<<1, 256, 0, stream>>>(w_ri, b_ri, w_rh, b_rh, w_ui, b_ui,
                                           w_uh, b_uh, w_oi, b_oi, w_oh, b_oh,
                                           WA, WB, bA, bB);
    for (int t = 0; t < CT; ++t) {
        convgru_step_ur<<<CB * (CH / 2), 256, 0, stream>>>(
            xT, hT, hrT, upd, bas, WA, bA, t);
        convgru_step_o<<<CB * (CH / 2), 256, 0, stream>>>(
            xT, hrT, upd, bas, WB, bB, hT, out, t);
    }
}
